// GPT_1915555414798
// MI455X (gfx1250) — compile-verified, matrix operand order checked by reading, not run on hardware
//
#include <hip/hip_runtime.h>
#include <hip/hip_bf16.h>
#include <math.h>

// ---------------- model constants ----------------
constexpr int  Bc  = 4;
constexpr int  Tc  = 1024;
constexpr int  Vc  = 32000;
constexpr int  Dc  = 768;
constexpr int  Hc  = 12;
constexpr int  Lc  = 6;
constexpr int  DHc = Dc / Hc;      // 64
constexpr int  FFc = 4 * Dc;       // 3072
constexpr int  Mc  = Bc * Tc;      // 4096 token rows

typedef __attribute__((ext_vector_type(16))) __bf16 v16bf;
typedef __attribute__((ext_vector_type(8)))  __bf16 v8bf;
typedef __attribute__((ext_vector_type(8)))  float  v8f;
typedef __attribute__((ext_vector_type(4)))  int    v4i;

#if __has_builtin(__builtin_amdgcn_global_load_async_to_lds_b128) && \
    __has_builtin(__builtin_amdgcn_s_wait_asynccnt)
#define USE_ASYNC_LDS 1
#else
#define USE_ASYNC_LDS 0
#endif

// Load a 16-element bf16 fragment for lane (lh = lane>>4 selects the k-phase):
// elements [0..7] = p[0..7], [8..15] = p[16..23]  (CDNA5 16-bit A/B layout)
__device__ __forceinline__ v16bf load_frag(const __bf16* p) {
    v8bf lo = *reinterpret_cast<const v8bf*>(p);
    v8bf hi = *reinterpret_cast<const v8bf*>(p + 16);
    v16bf r;
#pragma unroll
    for (int i = 0; i < 8; ++i) { r[i] = lo[i]; r[i + 8] = hi[i]; }
    return r;
}

__device__ __forceinline__ v8f wmma_bf16(v16bf a, v16bf b, v8f c) {
    return __builtin_amdgcn_wmma_f32_16x16x32_bf16(
        /*neg_a=*/false, a, /*neg_b=*/false, b,
        /*c_mod=*/(short)0, c, /*reuse_a=*/false, /*reuse_b=*/false);
}

// ---------------- f32 -> bf16 bulk convert ----------------
__global__ void cvt_bf16_kernel(const float* __restrict__ src,
                                __bf16* __restrict__ dst, long long n) {
    long long i = (long long)blockIdx.x * blockDim.x + threadIdx.x;
    long long stride = (long long)gridDim.x * blockDim.x;
    for (; i < n; i += stride) dst[i] = (__bf16)src[i];
}

// ---------------- embedding: x = tok_emb[idx] + pos + feather ----------------
__global__ void embed_kernel(const int* __restrict__ idx,
                             const float* __restrict__ feather,
                             const float* __restrict__ tok,
                             const float* __restrict__ pos,
                             float* __restrict__ x) {
    long long n = (long long)Mc * Dc;
    long long i = (long long)blockIdx.x * blockDim.x + threadIdx.x;
    long long stride = (long long)gridDim.x * blockDim.x;
    for (; i < n; i += stride) {
        int bt = (int)(i / Dc), d = (int)(i % Dc);
        int b = bt / Tc, t = bt % Tc;
        x[i] = tok[(long long)idx[bt] * Dc + d] + pos[(long long)t * Dc + d]
             + feather[(long long)b * Dc + d];
    }
}

// ---------------- layernorm (f32 in -> bf16 out) ----------------
__global__ void layernorm_kernel(const float* __restrict__ x,
                                 const float* __restrict__ g,
                                 const float* __restrict__ b,
                                 __bf16* __restrict__ out) {
    __shared__ float s1[256], s2[256];
    const int row = blockIdx.x, tid = threadIdx.x;
    const float* xr = x + (size_t)row * Dc;
    float a = 0.f, q = 0.f;
    for (int i = tid; i < Dc; i += 256) { float v = xr[i]; a += v; q += v * v; }
    s1[tid] = a; s2[tid] = q;
    __syncthreads();
    for (int off = 128; off > 0; off >>= 1) {
        if (tid < off) { s1[tid] += s1[tid + off]; s2[tid] += s2[tid + off]; }
        __syncthreads();
    }
    const float mu = s1[0] * (1.f / Dc);
    const float var = s2[0] * (1.f / Dc) - mu * mu;
    const float rstd = rsqrtf(var + 1e-5f);
    __bf16* orow = out + (size_t)row * Dc;
    for (int i = tid; i < Dc; i += 256)
        orow[i] = (__bf16)((xr[i] - mu) * rstd * g[i] + b[i]);
}

// ---------------- WMMA GEMM: Y = act(X[MxK] * W[NxK]^T + bias) (+resid) ------
// block = 256 threads = 8 waves; block tile 128 rows x 64 cols.
// Weight k-slabs (64x32 bf16 = 4KB) double-buffered in LDS, filled with
// async global->LDS copies (ASYNCcnt) and consumed by all 8 waves.
__global__ void gemm_bf16_kernel(const __bf16* __restrict__ X,
                                 const __bf16* __restrict__ W,
                                 const float* __restrict__ bias,
                                 const float* __restrict__ resid,
                                 float* __restrict__ outF,
                                 __bf16* __restrict__ outB,
                                 int M, int N, int K, int gelu) {
    __shared__ __bf16 lds_w[2][64][32];        // [buf][n-row][k]

    const int tid  = threadIdx.x;
    const int lane = tid & 31;
    const int wv   = tid >> 5;
    const int lh   = lane >> 4;      // k-phase (0/1)
    const int ll   = lane & 15;      // row/col within tile
    const int m0 = blockIdx.y * 128 + wv * 16;
    const int n0 = blockIdx.x * 64;

    // this thread's 16-byte chunk of the 64x32 weight slab
    const int wrow_ld = tid >> 2;            // 0..63
    const int wkc_ld  = (tid & 3) * 8;       // 0,8,16,24
    const __bf16* wsrc = W + (size_t)(n0 + wrow_ld) * K + wkc_ld;

    auto issue_slab = [&](int buf, int k0) {
#if USE_ASYNC_LDS
        __builtin_amdgcn_global_load_async_to_lds_b128(
            (__attribute__((address_space(1))) v4i*)(const_cast<__bf16*>(wsrc + k0)),
            (__attribute__((address_space(3))) v4i*)&lds_w[buf][wrow_ld][wkc_ld],
            0, 0);
#else
        v8bf v = *reinterpret_cast<const v8bf*>(wsrc + k0);
        *reinterpret_cast<v8bf*>(&lds_w[buf][wrow_ld][wkc_ld]) = v;
#endif
    };

    v8f acc[4] = {};
    const __bf16* xrow = X + (size_t)(m0 + ll) * K + lh * 8;

    issue_slab(0, 0);
    for (int k0 = 0; k0 < K; k0 += 32) {
        const int cur = (k0 >> 5) & 1;
        const bool more = (k0 + 32) < K;
        if (more) issue_slab(cur ^ 1, k0 + 32);   // prefetch next slab
#if USE_ASYNC_LDS
        if (more) {
            __builtin_amdgcn_s_wait_asynccnt(1);  // current slab landed, next in flight
        } else {
            __builtin_amdgcn_s_wait_asynccnt(0);  // final slab landed
        }
#endif
        __syncthreads();                          // slab visible to all waves

        v16bf a = load_frag(xrow + k0);           // per-wave A frag (global/L2)
#pragma unroll
        for (int t = 0; t < 4; ++t) {
            v16bf bt = load_frag(&lds_w[cur][t * 16 + ll][lh * 8]);
            acc[t] = wmma_bf16(a, bt, acc[t]);
        }
        __syncthreads();                          // all reads done before refill
    }

#pragma unroll
    for (int t = 0; t < 4; ++t) {
#pragma unroll
        for (int r = 0; r < 8; ++r) {
            int row = m0 + r + 8 * lh;
            int col = n0 + t * 16 + ll;
            float v = acc[t][r] + (bias ? bias[col] : 0.f);
            if (gelu) v = 0.5f * v * (1.f + erff(v * 0.70710678118654752f));
            size_t o = (size_t)row * N + col;
            if (resid) v += resid[o];
            if (outF) outF[o] = v;
            if (outB) outB[o] = (__bf16)v;
        }
    }
}

// ---------------- causal flash attention (streaming softmax) ----------------
// qkv: bf16 [B*T, 3*D]  (q | k | v).  yout: bf16 [B*T, D]
// grid = (T/64, B*H), block = 128 (4 waves); each wave owns a 16-row q tile.
__global__ void attn_kernel(const __bf16* __restrict__ qkv,
                            __bf16* __restrict__ yout) {
    __shared__ float  s_sc[4][16][32];
    __shared__ __bf16 s_p[4][16][32];
    __shared__ float  s_m[4][16], s_l[4][16], s_f[4][16];

    const int lane = threadIdx.x & 31;
    const int wv   = threadIdx.x >> 5;
    const int lh   = lane >> 4;
    const int ll   = lane & 15;
    const int bh = blockIdx.y, bb = bh / Hc, hh = bh % Hc;
    const int q0 = blockIdx.x * 64 + wv * 16;
    const size_t base = (size_t)bb * Tc * (3 * Dc);
    const int hoff = hh * DHc;

    // Q fragments: 16 rows x 64 dh -> two K=32 A-frags
    const __bf16* qp = qkv + base + (size_t)(q0 + ll) * (3 * Dc) + hoff + lh * 8;
    v16bf aq0 = load_frag(qp);
    v16bf aq1 = load_frag(qp + 32);

    v8f acc[4] = {};
    if (lane < 16) { s_m[wv][lane] = -1e30f; s_l[wv][lane] = 0.f; }

    const __bf16* kbase = qkv + base + Dc + hoff + lh * 8;       // K matrix
    const __bf16* vbase = qkv + base + 2 * Dc;                   // V matrix

    for (int j0 = 0; j0 < q0 + 16; j0 += 32) {
        // ---- S = Q K^T (two 16x16 n-tiles, K-depth 64 = 2 wmma each) ----
        v8f s[2] = {};
#pragma unroll
        for (int t = 0; t < 2; ++t) {
            const __bf16* kp = kbase + (size_t)(j0 + t * 16 + ll) * (3 * Dc);
            s[t] = wmma_bf16(aq0, load_frag(kp), s[t]);
            s[t] = wmma_bf16(aq1, load_frag(kp + 32), s[t]);
        }
#pragma unroll
        for (int r = 0; r < 8; ++r) {
            s_sc[wv][r + 8 * lh][ll]      = s[0][r] * 0.125f;    // 1/sqrt(64)
            s_sc[wv][r + 8 * lh][16 + ll] = s[1][r] * 0.125f;
        }

        // ---- online softmax (one lane per row; DS in-order within wave) ----
        if (lane < 16) {
            const int row_g = q0 + lane;
            float mo = s_m[wv][lane], lo = s_l[wv][lane];
            float sc[32], mx = mo;
#pragma unroll
            for (int c = 0; c < 32; ++c) {
                float v = (j0 + c <= row_g) ? s_sc[wv][lane][c] : -1e30f;
                sc[c] = v; mx = fmaxf(mx, v);
            }
            float f = __expf(mo - mx), ls = lo * f;
#pragma unroll
            for (int c = 0; c < 32; ++c) {
                float pv = __expf(sc[c] - mx);
                ls += pv;
                s_p[wv][lane][c] = (__bf16)pv;
            }
            s_m[wv][lane] = mx; s_l[wv][lane] = ls; s_f[wv][lane] = f;
        }

        // ---- rescale accumulators by row factors ----
        float fr[8];
#pragma unroll
        for (int r = 0; r < 8; ++r) fr[r] = s_f[wv][r + 8 * lh];
#pragma unroll
        for (int t = 0; t < 4; ++t)
#pragma unroll
            for (int r = 0; r < 8; ++r) acc[t][r] *= fr[r];

        // ---- P (16x32 bf16 A-frag from LDS) ----
        v16bf ap;
#pragma unroll
        for (int i = 0; i < 8; ++i) {
            ap[i]     = s_p[wv][ll][lh * 8 + i];
            ap[i + 8] = s_p[wv][ll][16 + lh * 8 + i];
        }

        // ---- acc += P * V  (4 dh tiles) ----
#pragma unroll
        for (int t = 0; t < 4; ++t) {
            const int col = hoff + t * 16 + ll;
            v16bf bv;
#pragma unroll
            for (int i = 0; i < 8; ++i) {
                bv[i]     = vbase[(size_t)(j0 + lh * 8 + i) * (3 * Dc) + col];
                bv[i + 8] = vbase[(size_t)(j0 + 16 + lh * 8 + i) * (3 * Dc) + col];
            }
            acc[t] = wmma_bf16(ap, bv, acc[t]);
        }
    }

    // ---- normalize and store ----
    if (lane < 16) s_f[wv][lane] = 1.f / s_l[wv][lane];
    float fr[8];
#pragma unroll
    for (int r = 0; r < 8; ++r) fr[r] = s_f[wv][r + 8 * lh];
#pragma unroll
    for (int t = 0; t < 4; ++t)
#pragma unroll
        for (int r = 0; r < 8; ++r) {
            int row = q0 + r + 8 * lh;
            int col = hoff + t * 16 + ll;
            yout[(size_t)(bb * Tc + row) * Dc + col] = (__bf16)(acc[t][r] * fr[r]);
        }
}

// ---------------- host-side orchestration ----------------
extern "C" void kernel_launch(void* const* d_in, const int* in_sizes, int n_in,
                              void* d_out, int out_size, void* d_ws, size_t ws_size,
                              hipStream_t stream) {
    (void)in_sizes; (void)n_in; (void)out_size; (void)ws_size;
    const int*   idx     = (const int*)  d_in[0];
    const float* feather = (const float*)d_in[1];
    const float* tok     = (const float*)d_in[2];
    const float* pos     = (const float*)d_in[3];
    const float* ln1g    = (const float*)d_in[4];
    const float* ln1b    = (const float*)d_in[5];
    const float* wqkv    = (const float*)d_in[6];
    const float* bqkv    = (const float*)d_in[7];
    const float* wproj   = (const float*)d_in[8];
    const float* bproj   = (const float*)d_in[9];
    const float* ln2g    = (const float*)d_in[10];
    const float* ln2b    = (const float*)d_in[11];
    const float* wfc1    = (const float*)d_in[12];
    const float* bfc1    = (const float*)d_in[13];
    const float* wfc2    = (const float*)d_in[14];
    const float* bfc2    = (const float*)d_in[15];
    const float* lnfg    = (const float*)d_in[16];
    const float* lnfb    = (const float*)d_in[17];
    const float* whead   = (const float*)d_in[18];

    // workspace carve (≈197 MB total)
    char* wp = (char*)d_ws;
    auto take = [&](size_t bytes) { void* r = wp; wp += ((bytes + 255) & ~(size_t)255); return r; };
    float*  x      = (float*) take((size_t)Mc * Dc * 4);
    __bf16* hbuf   = (__bf16*)take((size_t)Mc * Dc * 2);
    __bf16* qkvb   = (__bf16*)take((size_t)Mc * 3 * Dc * 2);
    __bf16* ffb    = (__bf16*)take((size_t)Mc * FFc * 2);
    __bf16* wqkvB  = (__bf16*)take((size_t)Lc * 3 * Dc * Dc * 2);
    __bf16* wprojB = (__bf16*)take((size_t)Lc * Dc * Dc * 2);
    __bf16* wfc1B  = (__bf16*)take((size_t)Lc * FFc * Dc * 2);
    __bf16* wfc2B  = (__bf16*)take((size_t)Lc * Dc * FFc * 2);
    __bf16* wheadB = (__bf16*)take((size_t)Vc * Dc * 2);

    // 1) weights -> bf16 (one streaming pass; amortized over 4096-row reuse)
    cvt_bf16_kernel<<<2048, 256, 0, stream>>>(wqkv,  wqkvB,  (long long)Lc * 3 * Dc * Dc);
    cvt_bf16_kernel<<<2048, 256, 0, stream>>>(wproj, wprojB, (long long)Lc * Dc * Dc);
    cvt_bf16_kernel<<<2048, 256, 0, stream>>>(wfc1,  wfc1B,  (long long)Lc * FFc * Dc);
    cvt_bf16_kernel<<<2048, 256, 0, stream>>>(wfc2,  wfc2B,  (long long)Lc * Dc * FFc);
    cvt_bf16_kernel<<<2048, 256, 0, stream>>>(whead, wheadB, (long long)Vc * Dc);

    // 2) embedding
    embed_kernel<<<4096, 256, 0, stream>>>(idx, feather, tok, pos, x);

    // 3) transformer layers
    for (int l = 0; l < Lc; ++l) {
        layernorm_kernel<<<Mc, 256, 0, stream>>>(x, ln1g + l * Dc, ln1b + l * Dc, hbuf);
        gemm_bf16_kernel<<<dim3(3 * Dc / 64, Mc / 128), 256, 0, stream>>>(
            hbuf, wqkvB + (size_t)l * 3 * Dc * Dc, bqkv + (size_t)l * 3 * Dc,
            nullptr, nullptr, qkvb, Mc, 3 * Dc, Dc, 0);
        attn_kernel<<<dim3(Tc / 64, Bc * Hc), 128, 0, stream>>>(qkvb, hbuf);
        gemm_bf16_kernel<<<dim3(Dc / 64, Mc / 128), 256, 0, stream>>>(
            hbuf, wprojB + (size_t)l * Dc * Dc, bproj + (size_t)l * Dc,
            x, x, nullptr, Mc, Dc, Dc, 0);
        layernorm_kernel<<<Mc, 256, 0, stream>>>(x, ln2g + l * Dc, ln2b + l * Dc, hbuf);
        gemm_bf16_kernel<<<dim3(FFc / 64, Mc / 128), 256, 0, stream>>>(
            hbuf, wfc1B + (size_t)l * FFc * Dc, bfc1 + (size_t)l * FFc,
            nullptr, nullptr, ffb, Mc, FFc, Dc, 1);
        gemm_bf16_kernel<<<dim3(Dc / 64, Mc / 128), 256, 0, stream>>>(
            ffb, wfc2B + (size_t)l * Dc * FFc, bfc2 + (size_t)l * Dc,
            x, x, nullptr, Mc, Dc, FFc, 0);
    }

    // 4) final LN + LM head (writes all of d_out, f32 logits [B,T,V])
    layernorm_kernel<<<Mc, 256, 0, stream>>>(x, lnfg, lnfb, hbuf);
    gemm_bf16_kernel<<<dim3(Vc / 64, Mc / 128), 256, 0, stream>>>(
        hbuf, wheadB, nullptr, nullptr, (float*)d_out, nullptr, Mc, Vc, Dc, 0);
}
